// BasicBlock_83150566850845
// MI455X (gfx1250) — 8-bit matrix operand order repaired and checked by reading, compiled, not run on hardware
//
#include <hip/hip_runtime.h>
#include <hip/hip_bf16.h>


typedef int v8i __attribute__((ext_vector_type(8)));

#define CONV_N 32
#define CONV_C 256
#define CONV_O 256
#define CONV_H 32
#define CONV_W 32
#define CONV_R 2304
#define NCHUNK 36
#define XR     48

__global__ __launch_bounds__(256) void binarize_kernel(
    const float* __restrict__ in, signed char* __restrict__ outb, int nelem)
{
  int idx = blockIdx.x * 256 + threadIdx.x;
  if (idx < nelem) outb[idx] = (in[idx] >= 0.0f) ? (signed char)1 : (signed char)-1;
}

__device__ __forceinline__ v8i load_afrag(const signed char* __restrict__ wp)
{
  int4 q0 = *(const int4*)(wp);
  int4 q1 = *(const int4*)(wp + 32);
  v8i a;
  a[0] = q0.x; a[1] = q0.y; a[2] = q0.z; a[3] = q0.w;
  a[4] = q1.x; a[5] = q1.y; a[6] = q1.z; a[7] = q1.w;
  return a;
}

__device__ __forceinline__ void build_btile(
    unsigned int* __restrict__ btbuf, const unsigned char* __restrict__ xs,
    const unsigned int* __restrict__ addrs, int tid)
{
  unsigned int wlo = 0u, whi = 0u;
  #pragma unroll
  for (int jj = 0; jj < 8; ++jj) {
    const unsigned int val = xs[addrs[jj]];
    if (jj < 4) wlo |= val << (8 * jj);
    else        whi |= val << (8 * (jj - 4));
  }
  btbuf[tid * 2]     = wlo;
  btbuf[tid * 2 + 1] = whi;
}

__device__ __forceinline__ void advance_addrs(
    unsigned int* __restrict__ addrs, unsigned int m)
{
  #pragma unroll
  for (int jj = 0; jj < 8; ++jj)
    addrs[jj] += 1009u + 45u * ((m >> jj) & 1u);
}

__global__ __launch_bounds__(256) void bconv_kernel(
    const signed char* __restrict__ xb,
    const signed char* __restrict__ wb,
    short* __restrict__ accout)
{
  __shared__ __attribute__((aligned(16))) unsigned char xs[CONV_C * 3 * XR];
  __shared__ __attribute__((aligned(32))) unsigned int  bt[2][2 * 32 * 8];

  const int tid = threadIdx.x;
  const int bx  = blockIdx.x;
  const int n   = bx >> 5;
  const int h   = bx & 31;

  {
    const int c = tid;
    for (int kh = 0; kh < 3; ++kh) {
      unsigned int ov[12];
      const int gr = h + kh - 1;
      if (gr >= 0 && gr < CONV_H) {
        const unsigned int* src = (const unsigned int*)
            (xb + (((size_t)n * CONV_C + c) * CONV_H + gr) * CONV_W);
        unsigned int a[8];
        #pragma unroll
        for (int k = 0; k < 8; ++k) a[k] = src[k];
        ov[0] = a[0] << 8;
        #pragma unroll
        for (int k = 1; k < 8; ++k) ov[k] = (a[k-1] >> 24) | (a[k] << 8);
        ov[8] = a[7] >> 24;
        ov[9] = 0u; ov[10] = 0u; ov[11] = 0u;
      } else {
        #pragma unroll
        for (int k = 0; k < 12; ++k) ov[k] = 0u;
      }
      unsigned int* dst = (unsigned int*)(xs + (c * 3 + kh) * XR);
      #pragma unroll
      for (int k = 0; k < 12; ++k) dst[k] = ov[k];
    }
  }

  const int lane = tid & 31;
  const int wv   = tid >> 5;
  const int wm   = wv & 3;
  const int wn   = wv >> 2;
  const int lhi  = lane >> 4;
  const int px   = wn * 16 + (lane & 15);

  const signed char* w0p =
      wb + (size_t)(wm * 64 + (lane & 15)) * CONV_R + (lhi << 4);
  const signed char* w1p = w0p + (size_t)16 * CONV_R;
  const signed char* w2p = w0p + (size_t)32 * CONV_R;
  const signed char* w3p = w0p + (size_t)48 * CONV_R;
  const int btoff = (wn * 32 + lane) * 8;

  unsigned int pat = 0u;
  #pragma unroll
  for (int k9 = 0; k9 < 9; ++k9) {
    const int kh = k9 / 3, kw = k9 - 3 * kh;
    const unsigned int bad = ((unsigned)(h + kh - 1) >= 32u) |
                             ((unsigned)(px + kw - 1) >= 32u);
    pat |= bad << k9;
  }
  const unsigned int par9 = pat ^ (0x1FFu * ((unsigned)__popc(pat) & 1u));

  int acc0[8], acc1[8], acc2[8], acc3[8];
  #pragma unroll
  for (int e = 0; e < 8; ++e) { acc0[e] = 0; acc1[e] = 0; acc2[e] = 0; acc3[e] = 0; }

  unsigned int addrs[8];
  unsigned int m0 = 0u, m1 = 0u, m2 = 0u;
  {
    const int vbase = tid * 8;
    const int wnb   = vbase >> 10;
    const int laneb = (vbase >> 5) & 31;
    const int j0    = vbase & 31;
    const int pb    = (laneb & 15) + wnb * 16;
    const int kadd  = (j0 & 16) + (laneb & 16);
    #pragma unroll
    for (int jj = 0; jj < 8; ++jj) {
      const int K  = j0 + jj + kadd;
      const int c  = K / 9;
      const int k9 = K - c * 9;
      const int kh = k9 / 3;
      const int kw = k9 - kh * 3;
      addrs[jj] = (unsigned)((c * 3 + kh) * XR + (pb + kw));
      m0 |= (unsigned)(kw == 2) << jj;
      m1 |= (unsigned)(kw == 1) << jj;
      m2 |= (unsigned)(kw == 0) << jj;
    }
  }

  __syncthreads();
  build_btile(bt[0], xs, addrs, tid);
  advance_addrs(addrs, m0);
  { unsigned int t = m0; m0 = m1; m1 = m2; m2 = t; }
  __syncthreads();

  int c9 = 0;
  for (int ch = 0; ch < NCHUNK; ++ch) {
    v8i bvec = *(const v8i*)&bt[ch & 1][btoff];
    v8i a0 = load_afrag(w0p + ch * 64);
    v8i a1 = load_afrag(w1p + ch * 64);
    v8i a2 = load_afrag(w2p + ch * 64);
    v8i a3 = load_afrag(w3p + ch * 64);
    __builtin_prefetch(w0p + ch * 64 + 64, 0, 3);
    __builtin_prefetch(w2p + ch * 64 + 64, 0, 3);

    v8i zero = {0, 0, 0, 0, 0, 0, 0, 0};
    v8i p0 = __builtin_amdgcn_wmma_i32_16x16x64_iu8(true, a0, true, bvec, zero, false, false);
    v8i p1 = __builtin_amdgcn_wmma_i32_16x16x64_iu8(true, a1, true, bvec, zero, false, false);
    v8i p2 = __builtin_amdgcn_wmma_i32_16x16x64_iu8(true, a2, true, bvec, zero, false, false);
    v8i p3 = __builtin_amdgcn_wmma_i32_16x16x64_iu8(true, a3, true, bvec, zero, false, false);

    if (ch + 1 < NCHUNK) {
      build_btile(bt[(ch + 1) & 1], xs, addrs, tid);
      advance_addrs(addrs, m0);
      { unsigned int t = m0; m0 = m1; m1 = m2; m2 = t; }
    }

    const int om = -(int)((par9 >> c9) & 1u);
    #pragma unroll
    for (int e = 0; e < 8; ++e) {
      int s0 = p0[e]; acc0[e] += s0 + (((s0 & 2) - 1) & om);
      int s1 = p1[e]; acc1[e] += s1 + (((s1 & 2) - 1) & om);
      int s2 = p2[e]; acc2[e] += s2 + (((s2 & 2) - 1) & om);
      int s3 = p3[e]; acc3[e] += s3 + (((s3 & 2) - 1) & om);
    }
    __syncthreads();
    c9 = (c9 == 8) ? 0 : (c9 + 1);
  }

  const int ob = wm * 64 + lhi * 8;
  short* outp = accout + (((size_t)n * CONV_O) * CONV_H + h) * CONV_W + px;
  #pragma unroll
  for (int e = 0; e < 8; ++e) {
    int v0 = acc0[e]; v0 = v0 < -254 ? -254 : (v0 > 254 ? 254 : v0);
    outp[(size_t)(ob + e)      * (CONV_H * CONV_W)] = (short)v0;
    int v1 = acc1[e]; v1 = v1 < -254 ? -254 : (v1 > 254 ? 254 : v1);
    outp[(size_t)(ob + 16 + e) * (CONV_H * CONV_W)] = (short)v1;
    int v2 = acc2[e]; v2 = v2 < -254 ? -254 : (v2 > 254 ? 254 : v2);
    outp[(size_t)(ob + 32 + e) * (CONV_H * CONV_W)] = (short)v2;
    int v3 = acc3[e]; v3 = v3 < -254 ? -254 : (v3 > 254 ? 254 : v3);
    outp[(size_t)(ob + 48 + e) * (CONV_H * CONV_W)] = (short)v3;
  }
}

__global__ __launch_bounds__(256) void stats_kernel(
    const short* __restrict__ acc, const float* __restrict__ resid,
    float* __restrict__ mean, float* __restrict__ var)
{
  __shared__ float ssum[256];
  __shared__ float ssq[256];
  const int o = blockIdx.x;
  const int t = threadIdx.x;
  float s = 0.0f, q = 0.0f;
  const bool has_res = (resid != nullptr);
  for (int idx = t; idx < CONV_N * 1024; idx += 256) {
    const int n  = idx >> 10;
    const int hw = idx & 1023;
    const size_t off = (((size_t)n * CONV_O + o) << 10) + hw;
    float v = (float)acc[off];
    if (has_res) v += resid[off];
    s += v; q += v * v;
  }
  ssum[t] = s; ssq[t] = q;
  __syncthreads();
  for (int st = 128; st > 0; st >>= 1) {
    if (t < st) { ssum[t] += ssum[t + st]; ssq[t] += ssq[t + st]; }
    __syncthreads();
  }
  if (t == 0) {
    float m = ssum[0] / 32768.0f;
    mean[o] = m;
    var[o]  = ssq[0] / 32768.0f - m * m;
  }
}

__global__ __launch_bounds__(256) void thresh_kernel(
    const short* __restrict__ acc, const float* __restrict__ mean,
    const float* __restrict__ var, const float* __restrict__ gamma,
    const float* __restrict__ beta, signed char* __restrict__ xb)
{
  int idx = blockIdx.x * 256 + threadIdx.x;
  if (idx >= CONV_N * CONV_O * 1024) return;
  int o = (idx >> 10) & 255;
  float rs = rsqrtf(var[o] + 1e-5f);
  float v  = ((float)acc[idx] - mean[o]) * rs * gamma[o] + beta[o];
  xb[idx] = (v >= 0.0f) ? (signed char)1 : (signed char)-1;
}

__global__ __launch_bounds__(256) void final_kernel(
    const short* __restrict__ acc, const float* __restrict__ x0,
    const float* __restrict__ mean, const float* __restrict__ var,
    const float* __restrict__ gamma, const float* __restrict__ beta,
    float* __restrict__ out)
{
  int idx = blockIdx.x * 256 + threadIdx.x;
  if (idx >= CONV_N * CONV_O * 1024) return;
  int o = (idx >> 10) & 255;
  float y  = (float)acc[idx] + x0[idx];
  float rs = rsqrtf(var[o] + 1e-5f);
  float v  = (y - mean[o]) * rs * gamma[o] + beta[o];
  v = v < -1.0f ? -1.0f : (v > 1.0f ? 1.0f : v);
  out[idx] = v;
}

__global__ __launch_bounds__(256) void reg_kernel(
    const float* __restrict__ reg0, float* __restrict__ out_reg)
{
  __shared__ int sred[256];
  const int t = threadIdx.x;
  int cnt = 0;
  for (int pp = t; pp < 1024; pp += 256) {
    const int h = pp >> 5, w = pp & 31;
    unsigned int pat = 0u;
    #pragma unroll
    for (int k9 = 0; k9 < 9; ++k9) {
      const int kh = k9 / 3, kw = k9 - 3 * kh;
      const unsigned int bad = ((unsigned)(h + kh - 1) >= 32u) |
                               ((unsigned)(w + kw - 1) >= 32u);
      pat |= bad << k9;
    }
    const unsigned int par9 = pat ^ (0x1FFu * ((unsigned)__popc(pat) & 1u));
    int c9 = 0;
    for (int chn = 0; chn < NCHUNK; ++chn) {
      cnt += (int)((par9 >> c9) & 1u);
      c9 = (c9 == 8) ? 0 : (c9 + 1);
    }
  }
  sred[t] = cnt;
  __syncthreads();
  for (int st = 128; st > 0; st >>= 1) {
    if (t < st) sred[t] += sred[t + st];
    __syncthreads();
  }
  if (t == 0) out_reg[0] = reg0[0] + 3.0f * ((float)sred[0] / 1024.0f);
}

extern "C" void kernel_launch(void* const* d_in, const int* in_sizes, int n_in,
                              void* d_out, int out_size, void* d_ws, size_t ws_size,
                              hipStream_t stream)
{
  const float* x0     = (const float*)d_in[0];
  const float* reg0   = (const float*)d_in[1];
  const float* w1     = (const float*)d_in[2];
  const float* gamma1 = (const float*)d_in[3];
  const float* beta1  = (const float*)d_in[4];
  const float* w2     = (const float*)d_in[5];
  const float* gamma2 = (const float*)d_in[6];
  const float* beta2  = (const float*)d_in[7];
  float* out = (float*)d_out;

  const int NELEM = CONV_N * CONV_O * 1024;
  const int WELEM = CONV_O * CONV_R;

  char* ws = (char*)d_ws;
  signed char* xb  = (signed char*)(ws);
  signed char* wb1 = (signed char*)(ws + 8388608);
  signed char* wb2 = wb1 + WELEM;
  short*       acc = (short*)(ws + 9568256);
  float*       m1  = (float*)(ws + 26345472);
  float*       v1  = m1 + 256;
  float*       m2  = v1 + 256;
  float*       v2  = m2 + 256;

  dim3 convGrid(CONV_N * CONV_H, 1, 1);

  binarize_kernel<<<NELEM / 256, 256, 0, stream>>>(x0, xb, NELEM);
  binarize_kernel<<<WELEM / 256, 256, 0, stream>>>(w1, wb1, WELEM);
  binarize_kernel<<<WELEM / 256, 256, 0, stream>>>(w2, wb2, WELEM);

  bconv_kernel<<<convGrid, 256, 0, stream>>>(xb, wb1, acc);
  stats_kernel<<<CONV_O, 256, 0, stream>>>(acc, nullptr, m1, v1);
  thresh_kernel<<<NELEM / 256, 256, 0, stream>>>(acc, m1, v1, gamma1, beta1, xb);

  bconv_kernel<<<convGrid, 256, 0, stream>>>(xb, wb2, acc);
  stats_kernel<<<CONV_O, 256, 0, stream>>>(acc, x0, m2, v2);
  final_kernel<<<NELEM / 256, 256, 0, stream>>>(acc, x0, m2, v2, gamma2, beta2, out);

  reg_kernel<<<1, 256, 0, stream>>>(reg0, out + NELEM);
}
